// SGC_28200755265752
// MI455X (gfx1250) — matrix operand order repaired and checked by reading, not compiled, not run on hardware
//
#include <hip/hip_runtime.h>
#include <cstdint>
#include <cstddef>

// ---------------- problem constants (match reference) ----------------
constexpr int N_NODES = 50000;
constexpr int N_EDGES = 800000;
constexpr int D_IN    = 512;
constexpr int D_HID   = 512;
constexpr int D_OUT   = 256;

#define DEV __device__ __forceinline__

// ---------------- WMMA vector types (gfx1250, wave32) ----------------
typedef __attribute__((ext_vector_type(16))) __bf16          v16bf;
typedef __attribute__((ext_vector_type(8)))  float           v8f;
typedef __attribute__((ext_vector_type(8)))  unsigned short  v8u;
typedef __attribute__((ext_vector_type(16))) unsigned short  v16u;

// ---------------- helpers ----------------
DEV unsigned short f2bf(float f) {              // f32 -> bf16, round-to-nearest-even
  unsigned u = __float_as_uint(f);
  u += 0x7FFFu + ((u >> 16) & 1u);
  return (unsigned short)(u >> 16);
}
DEV float bf2f(unsigned short h) { return __uint_as_float(((unsigned)h) << 16); }

DEV unsigned mix32(unsigned x) {                // deterministic dropout hash
  x ^= x >> 16; x *= 0x7feb352dU;
  x ^= x >> 15; x *= 0x846ca68bU;
  x ^= x >> 16; return x;
}

// ---------------- small elementwise kernels ----------------
__global__ void k_init_deg(int* deg) {          // self loop contributes 1
  int i = blockIdx.x * blockDim.x + threadIdx.x;
  if (i < N_NODES) deg[i] = 1;
}

__global__ void k_deg(const long long* __restrict__ ei, int* __restrict__ deg) {
  int e = blockIdx.x * blockDim.x + threadIdx.x;
  if (e < N_EDGES) atomicAdd(&deg[(int)ei[N_EDGES + e]], 1);
}

__global__ void k_dinv(const int* __restrict__ deg, float* __restrict__ dinv) {
  int i = blockIdx.x * blockDim.x + threadIdx.x;
  if (i < N_NODES) {
    float d = (float)deg[i];        // deg and dinv alias: read int, write float
    dinv[i] = rsqrtf(d);
  }
}

__global__ void k_f2b(const float* __restrict__ in, unsigned short* __restrict__ out, int n) {
  for (int i = blockIdx.x * blockDim.x + threadIdx.x; i < n; i += gridDim.x * blockDim.x)
    out[i] = f2bf(in[i]);
}

// agg[i,f] = dinv[i]^2 * x[i,f]   (self-loop term), f32 source
__global__ void k_selfinit_f32(const float* __restrict__ x, const float* __restrict__ dinv,
                               float* __restrict__ agg) {
  int i = blockIdx.x * blockDim.x + threadIdx.x;
  if (i < N_NODES * D_IN) {
    float dv = dinv[i >> 9];        // D = 512
    agg[i] = dv * dv * x[i];
  }
}
// bf16 source variant (embedding)
__global__ void k_selfinit_b16(const unsigned short* __restrict__ xb, const float* __restrict__ dinv,
                               float* __restrict__ agg) {
  int i = blockIdx.x * blockDim.x + threadIdx.x;
  if (i < N_NODES * D_HID) {
    float dv = dinv[i >> 9];
    agg[i] = dv * dv * bf2f(xb[i]);
  }
}

// ---------------- edge scatter: agg[dst] += norm * x[src] ----------------
// one block (128 threads) per edge; each thread covers 4 features (512 total)
__global__ __launch_bounds__(128)
void k_scatter_f32(const long long* __restrict__ ei, const float* __restrict__ dinv,
                   const float* __restrict__ x, float* __restrict__ agg) {
  __shared__ float s_norm;
  __shared__ int   s_src, s_dst;
  int e = blockIdx.x;
  if (threadIdx.x == 0) {
    int s = (int)ei[e];
    int d = (int)ei[N_EDGES + e];
    s_src = s; s_dst = d;
    s_norm = dinv[s] * dinv[d];
  }
  __syncthreads();
  int f = threadIdx.x * 4;
  float4 v = *(const float4*)(x + (size_t)s_src * D_IN + f);
  float* o = agg + (size_t)s_dst * D_IN + f;
  float nrm = s_norm;
  unsafeAtomicAdd(o + 0, nrm * v.x);
  unsafeAtomicAdd(o + 1, nrm * v.y);
  unsafeAtomicAdd(o + 2, nrm * v.z);
  unsafeAtomicAdd(o + 3, nrm * v.w);
}

__global__ __launch_bounds__(128)
void k_scatter_b16(const long long* __restrict__ ei, const float* __restrict__ dinv,
                   const unsigned short* __restrict__ xb, float* __restrict__ agg) {
  __shared__ float s_norm;
  __shared__ int   s_src, s_dst;
  int e = blockIdx.x;
  if (threadIdx.x == 0) {
    int s = (int)ei[e];
    int d = (int)ei[N_EDGES + e];
    s_src = s; s_dst = d;
    s_norm = dinv[s] * dinv[d];
  }
  __syncthreads();
  int f = threadIdx.x * 4;
  const unsigned short* p = xb + (size_t)s_src * D_HID + f;
  uint2 pk = *(const uint2*)p;                         // 4 bf16 in 8 bytes
  float v0 = bf2f((unsigned short)(pk.x & 0xFFFFu));
  float v1 = bf2f((unsigned short)(pk.x >> 16));
  float v2 = bf2f((unsigned short)(pk.y & 0xFFFFu));
  float v3 = bf2f((unsigned short)(pk.y >> 16));
  float* o = agg + (size_t)s_dst * D_HID + f;
  float nrm = s_norm;
  unsafeAtomicAdd(o + 0, nrm * v0);
  unsafeAtomicAdd(o + 1, nrm * v1);
  unsafeAtomicAdd(o + 2, nrm * v2);
  unsafeAtomicAdd(o + 3, nrm * v3);
}

// ---------------- WMMA GEMM: out = A(MxK,bf16) @ W(NOUTxK,bf16)^T + bias ----------------
// One block = one 16-row M tile. The A tile is a CONTIGUOUS 16*KC*2-byte block
// (tile spans the full K of 16 consecutive rows), staged into LDS once with
// global_load_async_to_lds_b128 (ASYNCcnt), then read via ds_load.
// Each wave32 owns a 16x64 output strip (4 accumulators); B fragments are
// pipelined one K-step ahead. The K loop is FULLY unrolled (KC is a template
// constant) so the bcur<-bnxt rotation is pure SSA renaming: no v_mov chains,
// no WMMA->VALU hazard nops at the back edge.
// FUSE=1: ReLU + hash-dropout(p=0.5, scale 2x) + also emit bf16 copy to outB.
template <int FUSE, int KC>
__global__ __launch_bounds__(256)
void k_gemm(const unsigned short* __restrict__ Ab,
            const unsigned short* __restrict__ Wb,
            const float* __restrict__ bias,
            float* __restrict__ outF,
            unsigned short* __restrict__ outB,
            int NOUT) {
  __shared__ unsigned short sA[16 * KC];     // 16 rows x KC bf16 (16 KB for KC=512)

  const int lane    = threadIdx.x & 31;
  const int wave    = threadIdx.x >> 5;
  const int nwaves  = blockDim.x >> 5;
  const int l15     = lane & 15;
  const int hi      = lane >> 4;             // 0: lanes 0-15, 1: lanes 16-31
  const int mbase   = blockIdx.x * 16;
  const int colbase = wave * 64;
  const int aoff    = hi * 16;               // A-fragment K-half select (contiguous k order)
  const int boff    = hi * 16;               // B-fragment K-half select (contiguous k order)

  // ---- stage A tile (contiguous 16*KC*2 bytes) into LDS via async copy ----
  {
    const char* gsrc  = (const char*)(Ab + (size_t)mbase * KC);
    unsigned    sbase = (unsigned)(uintptr_t)(&sA[0]);   // flat->LDS offset (addr[31:0])
    constexpr int CHUNKS = (16 * KC * 2) / 512;          // 512 B per wave-instruction
    for (int c = wave; c < CHUNKS; c += nwaves) {
      unsigned off  = (unsigned)(c * 512 + lane * 16);
      unsigned lofs = sbase + off;
      const char* g = gsrc + off;
      asm volatile("global_load_async_to_lds_b128 %0, %1, off"
                   :: "v"(lofs), "v"(g) : "memory");
    }
    asm volatile("s_wait_asynccnt 0" ::: "memory");      // this wave's copies done
    __syncthreads();                                     // all waves' copies visible
  }

  const unsigned short* sArow = sA + l15 * KC;

  // ---- per-tile B row pointers (W row == output column; contiguous in K) ----
  const unsigned short* Wcol[4];
#pragma unroll
  for (int t = 0; t < 4; ++t)
    Wcol[t] = Wb + (size_t)(colbase + t * 16 + l15) * KC + boff;

  v8f acc[4] = {v8f{}, v8f{}, v8f{}, v8f{}};

  // prologue: load B fragments for k0 = 0
  v16u bcur[4];
#pragma unroll
  for (int t = 0; t < 4; ++t) bcur[t] = *(const v16u*)(Wcol[t]);

#pragma unroll
  for (int k0 = 0; k0 < KC; k0 += 32) {
    const bool more = (k0 + 32 < KC);        // compile-time per unrolled iteration
    v16u bnxt[4];
    if (more) {
#pragma unroll
      for (int t = 0; t < 4; ++t) bnxt[t] = *(const v16u*)(Wcol[t] + k0 + 32);
    }

    // A fragment from LDS: row l15, k0+aoff..+15 (contiguous)
    v8u a0 = *(const v8u*)(sArow + k0 + aoff);
    v8u a1 = *(const v8u*)(sArow + k0 + aoff + 8);
    v16u au;
#pragma unroll
    for (int i = 0; i < 8; ++i) { au[i] = a0[i]; au[i + 8] = a1[i]; }
    v16bf a = __builtin_bit_cast(v16bf, au);

#pragma unroll
    for (int t = 0; t < 4; ++t) {
      v16bf b = __builtin_bit_cast(v16bf, bcur[t]);
      acc[t] = __builtin_amdgcn_wmma_f32_16x16x32_bf16(
          /*neg_a=*/false, a, /*neg_b=*/false, b,
          /*c_mod=*/(short)0, acc[t], /*reuse_a=*/false, /*reuse_b=*/false);
    }
    if (more) {
#pragma unroll
      for (int t = 0; t < 4; ++t) bcur[t] = bnxt[t];   // SSA rename after full unroll
    }
  }

  // Epilogue: C/D layout — VGPR r holds row hi*8+r, column l15
#pragma unroll
  for (int t = 0; t < 4; ++t) {
    int col = colbase + t * 16 + l15;
    float bs = bias[col];
#pragma unroll
    for (int r = 0; r < 8; ++r) {
      int m = mbase + hi * 8 + r;
      float v = acc[t][r] + bs;
      if (FUSE) {
        v = v > 0.0f ? v : 0.0f;                       // ReLU
        unsigned idx = (unsigned)(m * NOUT + col);
        v = (mix32(idx ^ 0x2Au) & 1u) ? v * 2.0f : 0.0f;  // dropout p=0.5
      }
      size_t o = (size_t)m * NOUT + col;
      outF[o] = v;
      if (FUSE) outB[o] = f2bf(v);
    }
  }
}

// ---------------- host-side orchestration ----------------
static inline size_t align256(size_t v) { return (v + 255) & ~(size_t)255; }

extern "C" void kernel_launch(void* const* d_in, const int* in_sizes, int n_in,
                              void* d_out, int out_size, void* d_ws, size_t ws_size,
                              hipStream_t stream) {
  (void)in_sizes; (void)n_in; (void)out_size; (void)ws_size;

  const float*     x  = (const float*)d_in[0];
  const long long* ei = (const long long*)d_in[1];     // int64 edge_index, shape (2,E)
  const float*     W1 = (const float*)d_in[2];         // (512,512)
  const float*     b1 = (const float*)d_in[3];
  const float*     W2 = (const float*)d_in[4];         // (256,512)
  const float*     b2 = (const float*)d_in[5];
  float*           out = (float*)d_out;                // [embedding N*512 | out N*256]

  char* ws = (char*)d_ws;
  size_t off = 0;
  float*          dinv = (float*)(ws + off);           off = align256(off + (size_t)N_NODES * 4);
  float*          aggF = (float*)(ws + off);           off = align256(off + (size_t)N_NODES * D_IN * 4);
  unsigned short* aggB = (unsigned short*)(ws + off);  off = align256(off + (size_t)N_NODES * D_IN * 2);
  unsigned short* embB = (unsigned short*)(ws + off);  off = align256(off + (size_t)N_NODES * D_HID * 2);
  unsigned short* w1B  = (unsigned short*)(ws + off);  off = align256(off + (size_t)D_HID * D_IN * 2);
  unsigned short* w2B  = (unsigned short*)(ws + off);  off = align256(off + (size_t)D_OUT * D_HID * 2);

  const int NB = 256;
  const int gN   = (N_NODES + NB - 1) / NB;
  const int gE   = (N_EDGES + NB - 1) / NB;
  const int gNF  = (N_NODES * D_IN + NB - 1) / NB;

  // --- GCN normalization: deg (with self loop) -> dinv ---
  k_init_deg<<<gN, NB, 0, stream>>>((int*)dinv);
  k_deg<<<gE, NB, 0, stream>>>(ei, (int*)dinv);
  k_dinv<<<gN, NB, 0, stream>>>((const int*)dinv, dinv);

  // --- weights -> bf16 ---
  k_f2b<<<(D_HID * D_IN + NB - 1) / NB, NB, 0, stream>>>(W1, w1B, D_HID * D_IN);
  k_f2b<<<(D_OUT * D_HID + NB - 1) / NB, NB, 0, stream>>>(W2, w2B, D_OUT * D_HID);

  // --- propagate 1: aggF = D^-1/2 (A+I) D^-1/2 x ---
  k_selfinit_f32<<<gNF, NB, 0, stream>>>(x, dinv, aggF);
  k_scatter_f32<<<N_EDGES, 128, 0, stream>>>(ei, dinv, x, aggF);
  k_f2b<<<gNF, NB, 0, stream>>>(aggF, aggB, N_NODES * D_IN);

  // --- GEMM1 (WMMA bf16): embedding = dropout(relu(agg @ W1^T + b1)) ---
  // 3125 M-tiles, 8 waves/block covering 512 columns
  k_gemm<1, 512><<<N_NODES / 16, 256, 0, stream>>>(aggB, w1B, b1, out, embB, D_HID);

  // --- propagate 2 on embedding (bf16 gather, f32 accumulate) ---
  k_selfinit_b16<<<gNF, NB, 0, stream>>>(embB, dinv, aggF);
  k_scatter_b16<<<N_EDGES, 128, 0, stream>>>(ei, dinv, embB, aggF);
  k_f2b<<<gNF, NB, 0, stream>>>(aggF, aggB, N_NODES * D_HID);

  // --- GEMM2 (WMMA bf16): out = agg2 @ W2^T + b2 ---
  // 3125 M-tiles, 4 waves/block covering 256 columns
  k_gemm<0, 512><<<N_NODES / 16, 128, 0, stream>>>(aggB, w2B, b2,
                                                   out + (size_t)N_NODES * D_HID,
                                                   nullptr, D_OUT);
}
